// SelfAttention_44444321579197
// MI455X (gfx1250) — matrix operand order repaired and checked by reading, not compiled, not run on hardware
//
#include <hip/hip_runtime.h>

// ---------------------------------------------------------------------------
// Types / helpers
// ---------------------------------------------------------------------------
typedef unsigned short u16;
typedef __attribute__((ext_vector_type(16))) __bf16 v16bf;
typedef __attribute__((ext_vector_type(8)))  float  v8f;
typedef __attribute__((ext_vector_type(4)))  unsigned int u32x4;
typedef __attribute__((ext_vector_type(8)))  int         i32x8;
typedef __attribute__((ext_vector_type(4)))  int         i32x4;

struct B32x8 { uint4 lo, hi; };

#if __has_builtin(__builtin_amdgcn_tensor_load_to_lds)
#define HAVE_TDM 1
#else
#define HAVE_TDM 0
#endif

__device__ __forceinline__ int lane_id() {
  // wave32: mbcnt_lo over full 32-bit mask == lane index
  return (int)__builtin_amdgcn_mbcnt_lo(~0u, 0u);
}

__device__ __forceinline__ u16 f2bf(float f) {  // round-to-nearest-even f32->bf16
  unsigned u = __float_as_uint(f);
  return (u16)((u + 0x7fffu + ((u >> 16) & 1u)) >> 16);
}

__device__ __forceinline__ void wait_tensorcnt0() {
#if __has_builtin(__builtin_amdgcn_s_wait_tensorcnt)
  __builtin_amdgcn_s_wait_tensorcnt(0);
#else
  asm volatile("s_wait_tensorcnt 0x0" ::: "memory");
#endif
}

#if HAVE_TDM
// Issue one TDM 2D tile load: bf16 tile [tile_h x tile_w] from a row-major
// tensor with row stride `row_stride` (elements) into LDS (packed row-major).
// Wave-uniform arguments; EXEC is ignored by TENSOR_LOAD_TO_LDS, so the caller
// must ensure only ONE wave executes this (one DMA per call).
// This toolchain's builtin is the 6-arg form:
//   (uint32x4 g0, int32x8 g1, int32x4 g2, int32x4 g3, int32x8 extra, i32 cpol)
__device__ __forceinline__ void tdm_load_2d_bf16(const u16* gsrc, void* ldst,
                                                 unsigned tensor_w, unsigned tensor_h,
                                                 unsigned tile_w, unsigned tile_h,
                                                 unsigned row_stride) {
  const unsigned long long ga = (unsigned long long)(size_t)gsrc;
  const unsigned la = (unsigned)(size_t)ldst;  // low 32 bits of generic ptr = LDS byte offset
  u32x4 g0;
  g0.x = 1u;                                   // count=1 valid user descriptor
  g0.y = la;                                   // lds_addr [63:32]
  g0.z = (unsigned)(ga & 0xffffffffu);         // global_addr [95:64]
  g0.w = (unsigned)((ga >> 32) & 0x1ffffffu)   // global_addr [120:96]
       | (2u << 30);                           // type=2 ("image") [127:126]
  i32x8 g1;
  g1[0] = (int)(1u << 16);                     // workgroup_mask=0, data_size=1 (2 bytes)
  g1[1] = (int)((tensor_w & 0xffffu) << 16);   // tensor_dim0 [15:0] -> bits 63:48
  g1[2] = (int)((tensor_w >> 16) | ((tensor_h & 0xffffu) << 16));  // dim0 hi | dim1 lo
  g1[3] = (int)((tensor_h >> 16) | ((tile_w & 0xffffu) << 16));    // dim1 hi | tile_dim0
  g1[4] = (int)(tile_h & 0xffffu);             // tile_dim1 ; tile_dim2 = 0
  g1[5] = (int)row_stride;                     // tensor_dim0_stride [31:0]
  g1[6] = 0;                                   // stride0 hi | tensor_dim1_stride lo
  g1[7] = 0;                                   // tensor_dim1_stride hi
  const i32x4 z4 = {0, 0, 0, 0};               // groups 2/3: 2D tensor -> unused
  const i32x8 z8 = {0, 0, 0, 0, 0, 0, 0, 0};   // trailing group: unused
  __builtin_amdgcn_tensor_load_to_lds(g0, g1, z4, z4, z8, 0);
}
#endif

// A operand, 16x32 bf16, source stored row-major [m][k] (stride in elements).
// Lane t: row = t&15 ; K pattern: [kb + kh, kb+kh+8) and [kb+16+kh, kb+16+kh+8), kh=(t>>4)*8
__device__ __forceinline__ v16bf load_a_frag(const u16* base, int stride, int kb) {
  const int t = lane_id();
  const u16* p = base + (size_t)(t & 15) * stride + kb + ((t >> 4) << 3);
  B32x8 r; r.lo = *(const uint4*)(p); r.hi = *(const uint4*)(p + 16);
  return __builtin_bit_cast(v16bf, r);
}

// B operand, 32x16 bf16, source stored as B^T row-major [n][k] (stride in elements).
// Lane t: row(n) = t&15 ; K pattern: [kb + kh, kb+kh+8) and [kb+16+kh, kb+16+kh+8), kh=(t>>4)*8
__device__ __forceinline__ v16bf load_b_frag(const u16* base, int stride, int kb) {
  const int t = lane_id();
  const u16* p = base + (size_t)(t & 15) * stride + kb + ((t >> 4) << 3);
  B32x8 r; r.lo = *(const uint4*)(p); r.hi = *(const uint4*)(p + 16);
  return __builtin_bit_cast(v16bf, r);
}

__device__ __forceinline__ v8f wmma_bf(v16bf a, v16bf b, v8f c) {
  return __builtin_amdgcn_wmma_f32_16x16x32_bf16(false, a, false, b, (short)0, c,
                                                 false, false);
}

// ---------------------------------------------------------------------------
// Problem constants:  B=32, L=128, HID=512, NH=8, D=64.  tokens = 4096.
// ---------------------------------------------------------------------------

// workspace layout (bytes)
static constexpr size_t OFF_WQT  = 0;                                  // [512][512] bf16 (W^T)
static constexpr size_t OFF_WKT  = OFF_WQT  + 512u * 512u * 2u;
static constexpr size_t OFF_WVT  = OFF_WKT  + 512u * 512u * 2u;
static constexpr size_t OFF_WSQT = OFF_WVT  + 512u * 512u * 2u;        // [64][512] bf16
static constexpr size_t OFF_WSKT = OFF_WSQT + 64u * 512u * 2u;
static constexpr size_t OFF_HSBF = OFF_WSKT + 64u * 512u * 2u;         // [4096][512] bf16
static constexpr size_t OFF_QBF  = OFF_HSBF + 4096u * 512u * 2u;       // [b][l][h][d] bf16
static constexpr size_t OFF_KBF  = OFF_QBF  + 4096u * 512u * 2u + 4096u; // [b][h][r][d] bf16 (+slack)
static constexpr size_t OFF_VTBF = OFF_KBF  + 4096u * 512u * 2u;       // [b][h][d][r] bf16
static constexpr size_t WS_NEED  = OFF_VTBF + 4096u * 512u * 2u;

// ---------------------------------------------------------------------------
// Prep kernels: bf16 conversions / transposes (tiny vs. main traffic)
// ---------------------------------------------------------------------------
__global__ __launch_bounds__(256) void k_prep_w(const float* __restrict__ Wq,
                                                const float* __restrict__ Wk,
                                                const float* __restrict__ Wv,
                                                u16* __restrict__ wqT,
                                                u16* __restrict__ wkT,
                                                u16* __restrict__ wvT) {
  const int i = blockIdx.x * 256 + threadIdx.x;       // 262144
  const int n = i & 511, k = i >> 9;
  const size_t src = (size_t)k * 512 + n, dst = (size_t)n * 512 + k;
  wqT[dst] = f2bf(Wq[src]);
  wkT[dst] = f2bf(Wk[src]);
  wvT[dst] = f2bf(Wv[src]);
}

__global__ __launch_bounds__(256) void k_prep_sw(const float* __restrict__ Wsq,
                                                 const float* __restrict__ Wsk,
                                                 u16* __restrict__ wsqT,
                                                 u16* __restrict__ wskT) {
  const int i = blockIdx.x * 256 + threadIdx.x;       // 32768
  const int n = i & 63, k = i >> 6;
  const size_t src = (size_t)k * 64 + n, dst = (size_t)n * 512 + k;
  wsqT[dst] = f2bf(Wsq[src]);
  wskT[dst] = f2bf(Wsk[src]);
}

__global__ __launch_bounds__(256) void k_prep_hs(const float* __restrict__ hs,
                                                 u16* __restrict__ hsbf) {
  const int i = blockIdx.x * 256 + threadIdx.x;       // 2097152
  hsbf[i] = f2bf(hs[i]);
}

// ---------------------------------------------------------------------------
// K1: QKV projection GEMM  (X[4096,512] @ W[512,512]) in bf16 WMMA.
// ---------------------------------------------------------------------------
__global__ __launch_bounds__(256) void k_qkv(const u16* __restrict__ hsbf,
                                             const u16* __restrict__ wqT,
                                             const u16* __restrict__ wkT,
                                             const u16* __restrict__ wvT,
                                             const float* __restrict__ bq,
                                             const float* __restrict__ bk,
                                             const float* __restrict__ bv,
                                             u16* __restrict__ qbf,
                                             u16* __restrict__ kbf,
                                             u16* __restrict__ vTbf) {
  const int mtile = blockIdx.x;           // 0..255
  const int mat   = blockIdx.y;           // 0=q 1=k 2=v
  const int wv    = threadIdx.x >> 5;
  const int t     = lane_id();
  const u16*  WT    = (mat == 0) ? wqT : (mat == 1) ? wkT : wvT;
  const float* bias = (mat == 0) ? bq : (mat == 1) ? bk : bv;

  v8f acc[4] = {};
  const u16* Abase = hsbf + (size_t)mtile * 16 * 512;
  for (int ks = 0; ks < 16; ++ks) {
    const int kb = ks * 32;
    v16bf a = load_a_frag(Abase, 512, kb);
#pragma unroll
    for (int dt = 0; dt < 4; ++dt) {
      const int ntile = wv * 4 + dt;
      acc[dt] = wmma_bf(a, load_b_frag(WT + (size_t)ntile * 16 * 512, 512, kb), acc[dt]);
    }
  }
  const int n0 = t & 15, hi = t >> 4;
#pragma unroll
  for (int dt = 0; dt < 4; ++dt) {
    const int n = (wv * 4 + dt) * 16 + n0;
    const float bb = bias[n];
    const int h = n >> 6, d = n & 63;
#pragma unroll
    for (int j = 0; j < 8; ++j) {
      const int token = mtile * 16 + j + hi * 8;
      const int b = token >> 7, r = token & 127;
      const u16 val = f2bf(acc[dt][j] + bb);
      if (mat == 0)      qbf[(size_t)token * 512 + n] = val;                        // [b][l][h][d]
      else if (mat == 1) kbf[((size_t)(b * 8 + h) * 128 + r) * 64 + d] = val;       // [b][h][r][d]
      else               vTbf[((size_t)(b * 8 + h) * 64 + d) * 128 + r] = val;      // [b][h][d][r]
    }
  }
}

// ---------------------------------------------------------------------------
// K2: fused structure projection + full score computation.  One WG per (b,l).
//   sq = sm[b,l] @ Wsq, sk = sm[b,l] @ Wsk  (never touch HBM)
//   scores[h,r] = q_l,h . sq_r  +  k_r,h . sk_r  +  q_l,h . k_r,h   -> /4 + mask
// Weight chunks staged by the Tensor Data Mover; sm chunk streamed + cvt bf16.
// ---------------------------------------------------------------------------
__global__ __launch_bounds__(256) void k_scores(const float* __restrict__ sm,
                                                const float* __restrict__ mask,
                                                const u16* __restrict__ wsqT,
                                                const u16* __restrict__ wskT,
                                                const float* __restrict__ bsq,
                                                const float* __restrict__ bsk,
                                                const u16* __restrict__ qbf,
                                                const u16* __restrict__ kbf,
                                                float* __restrict__ scoresOut) {
  __shared__ __align__(16) unsigned char ldsraw[40960];
  u16*   smc  = (u16*)(ldsraw);              // stage1: sm chunk  [128][64] bf16 (16KB)
  u16*   wsqc = (u16*)(ldsraw + 16384);      // stage1: WsqT chunk [64][64]  (8KB)
  u16*   wskc = (u16*)(ldsraw + 24576);      // stage1: WskT chunk [64][64]  (8KB)
  u16*   sql  = (u16*)(ldsraw);              // stage2: sq [128][64] bf16 (aliases smc)
  u16*   skl  = (u16*)(ldsraw + 16384);      // stage2: sk [128][64] bf16 (aliases w*)
  float* dsc  = (float*)(ldsraw + 32768);    // per-wave 16x16 diag scratch (8KB)

  const int bl = blockIdx.x;                 // 0..4095
  const int b = bl >> 7, l = bl & 127;
  const int tid = threadIdx.x;
  const int wv = tid >> 5;                   // wave -> r-strip [16*wv, 16*wv+16)
  const int t = lane_id();

  v8f sqa[4] = {}; v8f ska[4] = {};
  const float* smbase = sm + (size_t)bl * (128 * 512);
  const int srow = tid & 127, shalf = tid >> 7;

  for (int hc = 0; hc < 8; ++hc) {           // hid chunks of 64
    // prefetch next chunk of the dominant HBM stream
    if (hc + 1 < 8)
      __builtin_prefetch(smbase + (size_t)srow * 512 + (hc + 1) * 64 + shalf * 32, 0, 0);
    __syncthreads();
#if HAVE_TDM
    if (wv == 0) {  // one DMA per tile, issued by a single wave
      tdm_load_2d_bf16(wsqT + hc * 64, wsqc, 512u, 64u, 64u, 64u, 512u);
      tdm_load_2d_bf16(wskT + hc * 64, wskc, 512u, 64u, 64u, 64u, 512u);
    }
#else
    {  // manual weight-chunk staging
      const int row = tid >> 2, c0 = (tid & 3) * 16;
      const u16* s1 = wsqT + (size_t)row * 512 + hc * 64 + c0;
      const u16* s2 = wskT + (size_t)row * 512 + hc * 64 + c0;
      *(uint4*)(wsqc + row * 64 + c0)     = *(const uint4*)(s1);
      *(uint4*)(wsqc + row * 64 + c0 + 8) = *(const uint4*)(s1 + 8);
      *(uint4*)(wskc + row * 64 + c0)     = *(const uint4*)(s2);
      *(uint4*)(wskc + row * 64 + c0 + 8) = *(const uint4*)(s2 + 8);
    }
#endif
    {  // stream sm chunk f32 -> bf16 into LDS [r][h]
      const float* src = smbase + (size_t)srow * 512 + hc * 64 + shalf * 32;
      u16* dst = smc + srow * 64 + shalf * 32;
#pragma unroll
      for (int i = 0; i < 8; ++i) {
        const float4 v = *(const float4*)(src + i * 4);
        *(unsigned*)(dst + i * 4)     = (unsigned)f2bf(v.x) | ((unsigned)f2bf(v.y) << 16);
        *(unsigned*)(dst + i * 4 + 2) = (unsigned)f2bf(v.z) | ((unsigned)f2bf(v.w) << 16);
      }
    }
#if HAVE_TDM
    if (wv == 0) wait_tensorcnt0();          // TDM tiles resident before barrier
#endif
    __syncthreads();
    const u16* Ab = smc + (size_t)(wv * 16) * 64;
#pragma unroll
    for (int ks = 0; ks < 2; ++ks) {
      const int kb = ks * 32;
      v16bf a = load_a_frag(Ab, 64, kb);
#pragma unroll
      for (int dt = 0; dt < 4; ++dt) {
        sqa[dt] = wmma_bf(a, load_b_frag(wsqc + (size_t)(dt * 16) * 64, 64, kb), sqa[dt]);
        ska[dt] = wmma_bf(a, load_b_frag(wskc + (size_t)(dt * 16) * 64, 64, kb), ska[dt]);
      }
    }
  }
  __syncthreads();
  {  // + structure biases, spill sq/sk strips to LDS as bf16 [r][d]
    const int n0 = t & 15, hi = t >> 4;
#pragma unroll
    for (int dt = 0; dt < 4; ++dt) {
      const int d = dt * 16 + n0;
      const float b1 = bsq[d], b2 = bsk[d];
#pragma unroll
      for (int j = 0; j < 8; ++j) {
        const int r = wv * 16 + j + hi * 8;
        sql[r * 64 + d] = f2bf(sqa[dt][j] + b1);
        skl[r * 64 + d] = f2bf(ska[dt][j] + b2);
      }
    }
  }
  __syncthreads();
  {  // stage 2: scores for this wave's r-strip, all 8 heads
    const int n0 = t & 15, hi = t >> 4;
    const u16* qrow = qbf + (size_t)bl * 512;          // rows = heads, stride 64
    // term1: [16h x 16r] = q(16x64) @ sq_strip^T  (rows 8..15 are padding, never stored)
    v8f c1 = {};
#pragma unroll
    for (int ks = 0; ks < 2; ++ks) {
      const int kb = ks * 32;
      c1 = wmma_bf(load_a_frag(qrow, 64, kb),
                   load_b_frag(sql + (size_t)(wv * 16) * 64, 64, kb), c1);
    }
    const float mval = mask[b * 128 + wv * 16 + n0];
    float* dw = dsc + wv * 256;
#pragma unroll
    for (int h = 0; h < 8; ++h) {
      // diag( k_strip @ (sk_strip + q_h)^T ):  k.sk  +  k.q  per r
      v8f c2 = {};
      const u16* kbase = kbf + ((size_t)(b * 8 + h) * 128 + wv * 16) * 64;
      const u16* qh = qrow + h * 64;
#pragma unroll
      for (int ks = 0; ks < 2; ++ks) {
        const int kb = ks * 32;
        v16bf a = load_a_frag(kbase, 64, kb);
        c2 = wmma_bf(a, load_b_frag(skl + (size_t)(wv * 16) * 64, 64, kb), c2);
        c2 = wmma_bf(a, load_b_frag(qh, 0, kb), c2);   // broadcast q_h over all rows
      }
#pragma unroll
      for (int j = 0; j < 8; ++j) dw[(j + hi * 8) * 16 + n0] = c2[j];
      if (t < 16) {                                    // extract diagonal, emit scores
        const float dv = dw[t * 16 + t];
        scoresOut[((size_t)(b * 8 + h) * 128 + l) * 128 + wv * 16 + t] =
            (c1[h] + dv) * 0.25f + mval;
      }
    }
  }
}

// ---------------------------------------------------------------------------
// K3: per-(b,h) softmax + (probs @ v) via WMMA against v^T[d][r] bf16.
// ---------------------------------------------------------------------------
__global__ __launch_bounds__(256) void k_ctx(const float* __restrict__ scoresIn,
                                             const u16* __restrict__ vT,
                                             float* __restrict__ ctxOut) {
  __shared__ __align__(16) u16 probs[8][16 * 128];     // per-wave [l][r] bf16 (32KB)
  const int bh = blockIdx.x, b = bh >> 3, h = bh & 7;
  const int wv = threadIdx.x >> 5;
  const int t = lane_id();
  const int lrow = t & 15, half = t >> 4;
  const int l = wv * 16 + lrow;

  const float* srow = scoresIn + ((size_t)bh * 128 + l) * 128 + half * 64;
  float vals[64];
  float mx = -1e30f;
#pragma unroll
  for (int i = 0; i < 16; ++i) {
    const float4 v = *(const float4*)(srow + i * 4);
    vals[4 * i + 0] = v.x; vals[4 * i + 1] = v.y;
    vals[4 * i + 2] = v.z; vals[4 * i + 3] = v.w;
    mx = fmaxf(mx, fmaxf(fmaxf(v.x, v.y), fmaxf(v.z, v.w)));
  }
  mx = fmaxf(mx, __uint_as_float((unsigned)__builtin_amdgcn_ds_bpermute(
                    ((t ^ 16) << 2), (int)__float_as_uint(mx))));
  float sum = 0.f;
#pragma unroll
  for (int i = 0; i < 64; ++i) { vals[i] = __expf(vals[i] - mx); sum += vals[i]; }
  sum += __uint_as_float((unsigned)__builtin_amdgcn_ds_bpermute(
             ((t ^ 16) << 2), (int)__float_as_uint(sum)));
  const float inv = 1.0f / sum;

  u16* pr = &probs[wv][lrow * 128 + half * 64];
#pragma unroll
  for (int i = 0; i < 32; ++i)
    *(unsigned*)(pr + 2 * i) =
        (unsigned)f2bf(vals[2 * i] * inv) | ((unsigned)f2bf(vals[2 * i + 1] * inv) << 16);

  v8f acc[4] = {};
  const u16* Ab = &probs[wv][0];
  const u16* Vb = vT + (size_t)bh * (64 * 128);
#pragma unroll
  for (int ks = 0; ks < 4; ++ks) {
    const int kb = ks * 32;
    v16bf a = load_a_frag(Ab, 128, kb);
#pragma unroll
    for (int dt = 0; dt < 4; ++dt)
      acc[dt] = wmma_bf(a, load_b_frag(Vb + (size_t)(dt * 16) * 128, 128, kb), acc[dt]);
  }
#pragma unroll
  for (int dt = 0; dt < 4; ++dt) {
    const int d = dt * 16 + (t & 15);
#pragma unroll
    for (int j = 0; j < 8; ++j) {
      const int ll = wv * 16 + j + (t >> 4) * 8;
      ctxOut[((size_t)(b * 128 + ll)) * 512 + h * 64 + d] = acc[dt][j];
    }
  }
}

// ---------------------------------------------------------------------------
// Host launcher
// ---------------------------------------------------------------------------
extern "C" void kernel_launch(void* const* d_in, const int* in_sizes, int n_in,
                              void* d_out, int out_size, void* d_ws, size_t ws_size,
                              hipStream_t stream) {
  (void)in_sizes; (void)n_in; (void)out_size;
  const float* hs   = (const float*)d_in[0];
  const float* mask = (const float*)d_in[1];
  const float* sm   = (const float*)d_in[2];
  const float* Wq   = (const float*)d_in[3];
  const float* bq   = (const float*)d_in[4];
  const float* Wk   = (const float*)d_in[5];
  const float* bk   = (const float*)d_in[6];
  const float* Wv   = (const float*)d_in[7];
  const float* bv   = (const float*)d_in[8];
  const float* Wsq  = (const float*)d_in[9];
  const float* bsq  = (const float*)d_in[10];
  const float* Wsk  = (const float*)d_in[11];
  const float* bsk  = (const float*)d_in[12];

  if (ws_size < WS_NEED) return;  // workspace too small; nothing sane to do

  unsigned char* w = (unsigned char*)d_ws;
  u16* wqT  = (u16*)(w + OFF_WQT);
  u16* wkT  = (u16*)(w + OFF_WKT);
  u16* wvT  = (u16*)(w + OFF_WVT);
  u16* wsqT = (u16*)(w + OFF_WSQT);
  u16* wskT = (u16*)(w + OFF_WSKT);
  u16* hsbf = (u16*)(w + OFF_HSBF);
  u16* qbf  = (u16*)(w + OFF_QBF);
  u16* kbf  = (u16*)(w + OFF_KBF);
  u16* vTbf = (u16*)(w + OFF_VTBF);

  float* ctx    = (float*)d_out;                       // [32,128,512]
  float* scores = ctx + (size_t)32 * 128 * 512;        // [32,8,128,128]

  k_prep_w <<<dim3(1024),    dim3(256), 0, stream>>>(Wq, Wk, Wv, wqT, wkT, wvT);
  k_prep_sw<<<dim3(128),     dim3(256), 0, stream>>>(Wsq, Wsk, wsqT, wskT);
  k_prep_hs<<<dim3(8192),    dim3(256), 0, stream>>>(hs, hsbf);
  k_qkv    <<<dim3(256, 3),  dim3(256), 0, stream>>>(hsbf, wqT, wkT, wvT, bq, bk, bv,
                                                     qbf, kbf, vTbf);
  k_scores <<<dim3(4096),    dim3(256), 0, stream>>>(sm, mask, wsqT, wskT, bsq, bsk,
                                                     qbf, kbf, scores);
  k_ctx    <<<dim3(256),     dim3(256), 0, stream>>>(scores, vTbf, ctx);
}
